// Ex03_17437567222065
// MI455X (gfx1250) — matrix operand order repaired and checked by reading, not compiled, not run on hardware
//
#include <hip/hip_runtime.h>

// ---------------------------------------------------------------------------
// NCA step (LN -> 4x dilated depthwise conv -> LN -> MLP via bf16 WMMA) x 96
// MI455X / gfx1250: wave32, v_wmma_f32_16x16x32_bf16, big-LDS staging.
// LDS per block trimmed to 145 KB (h1/h2 tiles alias the dead u region) so
// two 4-wave blocks fit per 320 KB WGP -> 8 waves/WGP for latency hiding.
// ---------------------------------------------------------------------------

#define DEVINL __device__ __forceinline__

typedef __attribute__((ext_vector_type(16))) __bf16 v16bf;
typedef __attribute__((ext_vector_type(8)))  float  v8f;

static constexpr int CC   = 128;          // channels of x
static constexpr int OC   = 256;          // depthwise out channels per dilation
static constexpr int CAT  = 1152;         // 128 + 4*256
static constexpr int HID  = 64;
static constexpr int NPIX = 64 * 30 * 30; // 57600
static constexpr int ROUNDS = 96;

// padded LDS row strides (in bf16 elements) to avoid 64-bank conflicts
static constexpr int USTR  = 1160;        // 1152 + 8
static constexpr int HSTR  = 136;         // 128 + 8
static constexpr int H2STR = 72;          // 64 + 8

DEVINL unsigned short f2bf(float f) {
  unsigned u = __float_as_uint(f);
  unsigned r = u + 0x7fffu + ((u >> 16) & 1u);   // round-to-nearest-even
  return (unsigned short)(r >> 16);
}
DEVINL float bf2f(unsigned short s) { return __uint_as_float(((unsigned)s) << 16); }

DEVINL float wred(float v) {
#pragma unroll
  for (int o = 16; o > 0; o >>= 1) v += __shfl_xor(v, o, 32);
  return v;
}

DEVINL float gelu_exact(float x) {
  return 0.5f * x * (1.0f + erff(x * 0.7071067811865476f));
}

// A fragment, 16x32 bf16 (ISA 7.12.2): lanes 0-15 -> M=lane, K base 0;
// lanes 16-31 -> M=lane-16, K base +8. Halfs 0..7 -> K kb+0..7, 8..15 -> K kb+16..23.
DEVINL v16bf load_a_lds(const unsigned short* base, int lane, int kk, int stride) {
  int m  = lane & 15;
  int kb = (lane >> 4) << 3;
  const unsigned short* p = base + m * stride + kk + kb;
  union { v16bf v; uint4 q[2]; } f;
  f.q[0] = *(const uint4*)(p);        // K = kb+0 .. kb+7
  f.q[1] = *(const uint4*)(p + 16);   // K = kb+16 .. kb+23
  return f.v;
}

// B fragment, 32x16 bf16: lane -> column n = lane&15; halfs 0..7 -> K = (lane>>4)*8 + h, 8..15 -> K = 16 + (lane>>4)*8 + (h-8).
// Weights are pre-transposed to [N][K] bf16 so each lane reads two contiguous 16-byte runs (K kb+0..7, kb+16..23).
DEVINL v16bf load_b_gl(const unsigned short* wt, int lane, int kk, int stride, int t) {
  int n = (t << 4) + (lane & 15);
  const unsigned short* p = wt + n * stride + kk + ((lane >> 4) << 3);
  union { v16bf v; uint4 q[2]; } f;
  f.q[0] = *(const uint4*)(p);
  f.q[1] = *(const uint4*)(p + 16);
  return f.v;
}

DEVINL v8f wmma_bf16(v16bf a, v16bf b, v8f c) {
  return __builtin_amdgcn_wmma_f32_16x16x32_bf16(false, a, false, b, (short)0, c,
                                                 false, false);
}

// ---------------------------------------------------------------------------
// one-time: convert + transpose MLP weights to bf16 [N][K]
// ---------------------------------------------------------------------------
__global__ void k_prep(const float* __restrict__ w1, const float* __restrict__ w2,
                       const float* __restrict__ w3,
                       unsigned short* __restrict__ w1t,
                       unsigned short* __restrict__ w2t,
                       unsigned short* __restrict__ w3t) {
  int i = blockIdx.x * 256 + threadIdx.x;
  if (i < CAT * CC) {                    // w1: [1152][128] -> w1t[n][1152]
    int k = i / CC, n = i % CC;
    w1t[n * CAT + k] = f2bf(w1[i]);
    return;
  }
  i -= CAT * CC;
  if (i < CC * HID) {                    // w2: [128][64] -> w2t[n][128]
    int k = i / HID, n = i % HID;
    w2t[n * CC + k] = f2bf(w2[i]);
    return;
  }
  i -= CC * HID;
  if (i < HID * CC) {                    // w3: [64][128] -> w3t[n][64]
    int k = i / CC, n = i % CC;
    w3t[n * HID + k] = f2bf(w3[i]);
  }
}

// ---------------------------------------------------------------------------
// LN1: one wave per pixel (C=128 -> 4 ch/lane), write bf16 xn
// ---------------------------------------------------------------------------
__global__ __launch_bounds__(256) void k_ln1(const float* __restrict__ x,
                                             const float* __restrict__ g,
                                             const float* __restrict__ b,
                                             unsigned short* __restrict__ xn) {
  int wid = threadIdx.x >> 5, lane = threadIdx.x & 31;
  int p = blockIdx.x * 8 + wid;
  const float4 xv = *(const float4*)(x + (size_t)p * CC + lane * 4);
  float s  = xv.x + xv.y + xv.z + xv.w;
  float sq = xv.x * xv.x + xv.y * xv.y + xv.z * xv.z + xv.w * xv.w;
  s = wred(s); sq = wred(sq);
  float mu = s * (1.0f / CC);
  float rs = rsqrtf(fmaxf(sq * (1.0f / CC) - mu * mu, 0.0f) + 1e-5f);
  const float4 gv = *(const float4*)(g + lane * 4);
  const float4 bv = *(const float4*)(b + lane * 4);
  float o0 = (xv.x - mu) * rs * gv.x + bv.x;
  float o1 = (xv.y - mu) * rs * gv.y + bv.y;
  float o2 = (xv.z - mu) * rs * gv.z + bv.z;
  float o3 = (xv.w - mu) * rs * gv.w + bv.w;
  uint2 pk;
  pk.x = (unsigned)f2bf(o0) | ((unsigned)f2bf(o1) << 16);
  pk.y = (unsigned)f2bf(o2) | ((unsigned)f2bf(o3) << 16);
  *(uint2*)(xn + (size_t)p * CC + lane * 4) = pk;
}

// ---------------------------------------------------------------------------
// fused step: depthwise convs -> u in LDS -> LN2 -> 3-layer MLP (WMMA) -> residual
// 128 threads = 4 waves; each wave owns 16 pixels (fully wave-local, no barriers).
// h1/h2 staging tiles alias the wave's own (dead) u region: per-wave LDS ops are
// hardware in-order, and explicit s_wait_dscnt fences separate the phases.
// ---------------------------------------------------------------------------
__global__ __launch_bounds__(128) void k_step(
    const float* __restrict__ xcur, const unsigned short* __restrict__ xn,
    const float* __restrict__ cw, const float* __restrict__ ln2g,
    const float* __restrict__ ln2b, const unsigned short* __restrict__ w1t,
    const float* __restrict__ b1, const unsigned short* __restrict__ w2t,
    const float* __restrict__ b2, const unsigned short* __restrict__ w3t,
    const float* __restrict__ b3, const float* __restrict__ ls,
    float* __restrict__ xnew) {
  extern __shared__ unsigned short smem[];
  const int wv = threadIdx.x >> 5, lane = threadIdx.x & 31;

  unsigned short* urow  = smem + wv * 16 * USTR;   // 16 x 1152 (u, layer-1 A)
  unsigned short* hbuf  = urow;                    // 16 x 128, aliases dead u
  unsigned short* h2buf = urow + 16 * HSTR;        // 16 x 64,  disjoint from hbuf

  const int pbase = blockIdx.x * 64 + wv * 16;

  // warm L2/WGP$ for the streamed weight matrices (global_prefetch_b8)
  __builtin_prefetch(w1t + (size_t)threadIdx.x * 1152, 0, 1);
  __builtin_prefetch(w1t + (size_t)threadIdx.x * 1152 + 576, 0, 1);
  __builtin_prefetch(w2t + (size_t)threadIdx.x * 64, 0, 1);
  __builtin_prefetch(w3t + (size_t)threadIdx.x * 64, 0, 1);

  // ---- phase 1: build u = [xn | conv_d1 | conv_d2 | conv_d4 | conv_d8], LN2 ----
#pragma unroll 1
  for (int m = 0; m < 16; ++m) {
    int p = pbase + m;
    int bb = p / 900, rem = p - bb * 900;
    int h = rem / 30, w = rem - h * 30;
    const size_t prow = (size_t)p * CC;
    float s = 0.0f, sq = 0.0f;

    { // xn passthrough (4 channels per lane)
      uint2 pk = *(const uint2*)(xn + prow + lane * 4);
      float v0 = bf2f((unsigned short)(pk.x & 0xffff));
      float v1 = bf2f((unsigned short)(pk.x >> 16));
      float v2 = bf2f((unsigned short)(pk.y & 0xffff));
      float v3 = bf2f((unsigned short)(pk.y >> 16));
      s += v0 + v1 + v2 + v3;
      sq += v0 * v0 + v1 * v1 + v2 * v2 + v3 * v3;
      *(uint2*)(urow + m * USTR + lane * 4) = pk;
    }

    // depthwise dilated 3x3, multiplier 2: ic -> oc {2ic, 2ic+1}
#pragma unroll
    for (int d = 0; d < 4; ++d) {
      const int dil = 1 << d;
#pragma unroll
      for (int ii = 0; ii < 4; ++ii) {
        const int ic = lane + 32 * ii;
        float a0 = 0.0f, a1 = 0.0f;
#pragma unroll
        for (int ky = 0; ky < 3; ++ky) {
          int hh = h + (ky - 1) * dil;
#pragma unroll
          for (int kx = 0; kx < 3; ++kx) {
            int ww = w + (kx - 1) * dil;
            if ((unsigned)hh < 30u && (unsigned)ww < 30u) {
              float xv = bf2f(xn[((size_t)(bb * 900 + hh * 30 + ww)) * CC + ic]);
              const float* wp = cw + (((d * 3 + ky) * 3 + kx) * OC + 2 * ic);
              a0 = fmaf(xv, wp[0], a0);
              a1 = fmaf(xv, wp[1], a1);
            }
          }
        }
        int c = CC + d * OC + 2 * ic;
        urow[m * USTR + c]     = f2bf(a0);
        urow[m * USTR + c + 1] = f2bf(a1);
        s += a0 + a1;
        sq += a0 * a0 + a1 * a1;
      }
    }

    // wave-local LN2 over 1152 and in-place normalize of row m
    s = wred(s); sq = wred(sq);
    float mu = s * (1.0f / CAT);
    float rs = rsqrtf(fmaxf(sq * (1.0f / CAT) - mu * mu, 0.0f) + 1e-5f);
    asm volatile("s_wait_dscnt 0" ::: "memory");   // same-wave LDS write->read order
#pragma unroll 4
    for (int i = 0; i < 36; ++i) {
      int c = lane * 36 + i;
      float v = bf2f(urow[m * USTR + c]);
      v = (v - mu) * rs * ln2g[c] + ln2b[c];
      urow[m * USTR + c] = f2bf(v);
    }
  }
  asm volatile("s_wait_dscnt 0" ::: "memory");

  const int mrow = (lane >> 4) << 3;   // 0 or 8
  const int ncol = lane & 15;
  const v8f vzero = {0.f, 0.f, 0.f, 0.f, 0.f, 0.f, 0.f, 0.f};

  // ---- layer 1: [16 x 1152] x [1152 x 128] ----
  v8f acc[8];
#pragma unroll
  for (int t = 0; t < 8; ++t) acc[t] = vzero;
  for (int kk = 0; kk < CAT; kk += 32) {
    v16bf a = load_a_lds(urow, lane, kk, USTR);
#pragma unroll
    for (int t = 0; t < 8; ++t)
      acc[t] = wmma_bf16(a, load_b_gl(w1t, lane, kk, CAT, t), acc[t]);
  }
  asm volatile("s_wait_dscnt 0" ::: "memory");     // u reads done; hbuf may alias
#pragma unroll
  for (int t = 0; t < 8; ++t) {
    int n = t * 16 + ncol;
    float bias = b1[n];
#pragma unroll
    for (int r = 0; r < 8; ++r)
      hbuf[(mrow + r) * HSTR + n] = f2bf(gelu_exact(acc[t][r] + bias));
  }
  asm volatile("s_wait_dscnt 0" ::: "memory");

  // ---- layer 2: [16 x 128] x [128 x 64] ----
  v8f acc2[4];
#pragma unroll
  for (int t = 0; t < 4; ++t) acc2[t] = vzero;
#pragma unroll
  for (int kk = 0; kk < CC; kk += 32) {
    v16bf a = load_a_lds(hbuf, lane, kk, HSTR);
#pragma unroll
    for (int t = 0; t < 4; ++t)
      acc2[t] = wmma_bf16(a, load_b_gl(w2t, lane, kk, CC, t), acc2[t]);
  }
#pragma unroll
  for (int t = 0; t < 4; ++t) {
    int n = t * 16 + ncol;
    float bias = b2[n];
#pragma unroll
    for (int r = 0; r < 8; ++r)
      h2buf[(mrow + r) * H2STR + n] = f2bf(gelu_exact(acc2[t][r] + bias));
  }
  asm volatile("s_wait_dscnt 0" ::: "memory");

  // ---- layer 3: [16 x 64] x [64 x 128] ----
  v8f acc3[8];
#pragma unroll
  for (int t = 0; t < 8; ++t) acc3[t] = vzero;
#pragma unroll
  for (int kk = 0; kk < HID; kk += 32) {
    v16bf a = load_a_lds(h2buf, lane, kk, H2STR);
#pragma unroll
    for (int t = 0; t < 8; ++t)
      acc3[t] = wmma_bf16(a, load_b_gl(w3t, lane, kk, HID, t), acc3[t]);
  }

  // ---- epilogue: x_next = x + (h @ w3 + b3) * ls ----
#pragma unroll
  for (int t = 0; t < 8; ++t) {
    int c = t * 16 + ncol;
    float b3v = b3[c], lsv = ls[c];
#pragma unroll
    for (int r = 0; r < 8; ++r) {
      size_t idx = (size_t)(pbase + mrow + r) * CC + c;
      xnew[idx] = xcur[idx] + (acc3[t][r] + b3v) * lsv;
    }
  }
}

// ---------------------------------------------------------------------------
// final: LN(x) @ wo + bo  (one wave per pixel, 11 outputs)
// ---------------------------------------------------------------------------
__global__ __launch_bounds__(256) void k_out(const float* __restrict__ x,
                                             const float* __restrict__ g,
                                             const float* __restrict__ b,
                                             const float* __restrict__ wo,
                                             const float* __restrict__ bo,
                                             float* __restrict__ out) {
  int wid = threadIdx.x >> 5, lane = threadIdx.x & 31;
  int p = blockIdx.x * 8 + wid;
  const float4 xv = *(const float4*)(x + (size_t)p * CC + lane * 4);
  float s  = xv.x + xv.y + xv.z + xv.w;
  float sq = xv.x * xv.x + xv.y * xv.y + xv.z * xv.z + xv.w * xv.w;
  s = wred(s); sq = wred(sq);
  float mu = s * (1.0f / CC);
  float rs = rsqrtf(fmaxf(sq * (1.0f / CC) - mu * mu, 0.0f) + 1e-5f);
  const float4 gv = *(const float4*)(g + lane * 4);
  const float4 bv = *(const float4*)(b + lane * 4);
  float n0 = (xv.x - mu) * rs * gv.x + bv.x;
  float n1 = (xv.y - mu) * rs * gv.y + bv.y;
  float n2 = (xv.z - mu) * rs * gv.z + bv.z;
  float n3 = (xv.w - mu) * rs * gv.w + bv.w;
  int c0 = lane * 4;
#pragma unroll
  for (int j = 0; j < 11; ++j) {
    float part = n0 * wo[(c0 + 0) * 11 + j] + n1 * wo[(c0 + 1) * 11 + j] +
                 n2 * wo[(c0 + 2) * 11 + j] + n3 * wo[(c0 + 3) * 11 + j];
    part = wred(part);
    if (lane == 0) out[(size_t)p * 11 + j] = part + bo[j];
  }
}

// ---------------------------------------------------------------------------
extern "C" void kernel_launch(void* const* d_in, const int* in_sizes, int n_in,
                              void* d_out, int out_size, void* d_ws, size_t ws_size,
                              hipStream_t stream) {
  const float* x    = (const float*)d_in[0];
  const float* ln1g = (const float*)d_in[1];
  const float* ln1b = (const float*)d_in[2];
  const float* cw   = (const float*)d_in[3];
  const float* ln2g = (const float*)d_in[4];
  const float* ln2b = (const float*)d_in[5];
  const float* w1   = (const float*)d_in[6];
  const float* b1   = (const float*)d_in[7];
  const float* w2   = (const float*)d_in[8];
  const float* b2   = (const float*)d_in[9];
  const float* w3   = (const float*)d_in[10];
  const float* b3   = (const float*)d_in[11];
  const float* ls   = (const float*)d_in[12];
  const float* outg = (const float*)d_in[13];
  const float* outb = (const float*)d_in[14];
  const float* wo   = (const float*)d_in[15];
  const float* bo   = (const float*)d_in[16];

  char* ws = (char*)d_ws;
  float* bufA = (float*)ws;                  ws += (size_t)NPIX * CC * sizeof(float);
  float* bufB = (float*)ws;                  ws += (size_t)NPIX * CC * sizeof(float);
  unsigned short* xnb = (unsigned short*)ws; ws += (size_t)NPIX * CC * 2;
  unsigned short* w1t = (unsigned short*)ws; ws += (size_t)CC * CAT * 2;
  unsigned short* w2t = (unsigned short*)ws; ws += (size_t)HID * CC * 2;
  unsigned short* w3t = (unsigned short*)ws; ws += (size_t)CC * HID * 2;

  k_prep<<<640, 256, 0, stream>>>(w1, w2, w3, w1t, w2t, w3t);

  const size_t smem = (size_t)(64 * USTR) * 2;   // 148,480 B -> 2 blocks / WGP
  const float* src = x;
  float* bufs[2] = {bufA, bufB};
  for (int r = 0; r < ROUNDS; ++r) {
    k_ln1<<<NPIX / 8, 256, 0, stream>>>(src, ln1g, ln1b, xnb);
    float* dst = bufs[r & 1];
    k_step<<<NPIX / 64, 128, smem, stream>>>(src, xnb, cw, ln2g, ln2b, w1t, b1,
                                             w2t, b2, w3t, b3, ls, dst);
    src = dst;
  }
  k_out<<<NPIX / 8, 256, 0, stream>>>(src, outg, outb, wo, bo, (float*)d_out);
}
